// PointNetMSGDown_14482629722279
// MI455X (gfx1250) — matrix operand order repaired and checked by reading, not compiled, not run on hardware
//
#include <hip/hip_runtime.h>
#include <hip/hip_bf16.h>

typedef __attribute__((ext_vector_type(16))) _Float16 v16h;
typedef __attribute__((ext_vector_type(8)))  _Float16 v8h;
typedef __attribute__((ext_vector_type(8)))  float    v8f;

#define N0_PTS   16384
#define SPOINT   4096
#define BATCH    4
#define DIN      32

// ---------------------------------------------------------------------------
// 1) Furthest point sampling: 1 block (1024 thr) per batch, points + running
//    dists resident in VGPRs; packed u64 (distbits<<32 | ~idx) max-reduce.
// ---------------------------------------------------------------------------
__global__ __launch_bounds__(1024) void fps_kernel(const float* __restrict__ qp,
                                                   int* __restrict__ sel) {
  const int b   = blockIdx.x;
  const int tid = threadIdx.x;
  const float* px = qp + (size_t)b * 3 * N0_PTS;
  const float* py = px + N0_PTS;
  const float* pz = px + 2 * N0_PTS;

  const int PPT = N0_PTS / 1024;   // 16 points per thread
  float X[16], Y[16], Z[16], Dst[16];
#pragma unroll
  for (int u = 0; u < PPT; ++u) {
    int i = u * 1024 + tid;
    X[u] = px[i]; Y[u] = py[i]; Z[u] = pz[i];
    Dst[u] = 1e10f;
  }

  __shared__ unsigned long long red[32];
  __shared__ int lastIdx;
  if (tid == 0) { lastIdx = 0; sel[b * SPOINT] = 0; }
  __syncthreads();

  const int lane = tid & 31, w = tid >> 5;
  for (int t = 1; t < SPOINT; ++t) {
    int li = lastIdx;
    float lx = px[li], ly = py[li], lz = pz[li];   // broadcast loads (L0/L2)
    unsigned long long best = 0ull;
#pragma unroll
    for (int u = 0; u < PPT; ++u) {
      float dx = X[u] - lx, dy = Y[u] - ly, dz = Z[u] - lz;
      float d  = dx * dx + dy * dy + dz * dz;
      float nd = fminf(Dst[u], d);
      Dst[u] = nd;
      unsigned long long key =
          ((unsigned long long)__float_as_uint(nd) << 32) |
          (unsigned int)(~(unsigned int)(u * 1024 + tid));
      best = (key > best) ? key : best;
    }
#pragma unroll
    for (int off = 16; off; off >>= 1) {
      unsigned long long o = __shfl_xor(best, off, 32);
      best = (o > best) ? o : best;
    }
    if (lane == 0) red[w] = best;
    __syncthreads();
    if (w == 0) {
      unsigned long long v = red[lane];
#pragma unroll
      for (int off = 16; off; off >>= 1) {
        unsigned long long o = __shfl_xor(v, off, 32);
        v = (o > v) ? o : v;
      }
      if (lane == 0) {
        int win = (int)(~(unsigned int)(v & 0xffffffffu));
        lastIdx = win;
        sel[b * SPOINT + t] = win;
      }
    }
    __syncthreads();
  }
}

// ---------------------------------------------------------------------------
// 2) Gather sampled xyz into d_out[0 .. B*3*S)
// ---------------------------------------------------------------------------
__global__ void gather_xyz_kernel(const float* __restrict__ qp,
                                  const int* __restrict__ sel,
                                  float* __restrict__ out) {
  int tid = blockIdx.x * 256 + threadIdx.x;          // over B*S
  int b = tid >> 12, s = tid & (SPOINT - 1);
  int idx = sel[tid];
#pragma unroll
  for (int c = 0; c < 3; ++c)
    out[((size_t)b * 3 + c) * SPOINT + s] = qp[((size_t)b * 3 + c) * N0_PTS + idx];
}

// ---------------------------------------------------------------------------
// 3) Ball query: wave per center; ordered scan, first-k in-radius indices,
//    pad with first hit (or 0 if none). Early exit when k found.
// ---------------------------------------------------------------------------
__global__ __launch_bounds__(256) void ball_query_kernel(
    const float* __restrict__ sp, const float* __restrict__ xyz,
    int* __restrict__ nidx, float r2, int k) {
  const int lane = threadIdx.x & 31;
  const int w    = threadIdx.x >> 5;
  const int gs   = blockIdx.x * 8 + w;               // over B*S
  const int b = gs >> 12, s = gs & (SPOINT - 1);
  const float* sx = sp + (size_t)b * 3 * N0_PTS;
  const float cx = xyz[((size_t)b * 3 + 0) * SPOINT + s];
  const float cy = xyz[((size_t)b * 3 + 1) * SPOINT + s];
  const float cz = xyz[((size_t)b * 3 + 2) * SPOINT + s];
  int* out = nidx + (size_t)gs * k;

  int count = 0, firstN = 0;
  for (int n0 = 0; n0 < N0_PTS && count < k; n0 += 32) {
    int n = n0 + lane;
    float dx = sx[n] - cx;
    float dy = sx[N0_PTS + n] - cy;
    float dz = sx[2 * N0_PTS + n] - cz;
    bool in = (dx * dx + dy * dy + dz * dz) < r2;
    unsigned int mask = (unsigned int)__ballot(in);
    if (count == 0 && mask) firstN = n0 + __ffs(mask) - 1;
    if (in) {
      int pos = count + __popc(mask & ((1u << lane) - 1u));
      if (pos < k) out[pos] = n;
    }
    count += __popc(mask);
  }
  if (lane < k && lane >= count) out[lane] = firstN;  // firstN==0 when none
}

// ---------------------------------------------------------------------------
// 4) Gather grouped features -> f16 activation matrix [M][32], M = B*S*k
// ---------------------------------------------------------------------------
__global__ void gather_feats_kernel(const float* __restrict__ x,
                                    const int* __restrict__ nidx,
                                    _Float16* __restrict__ Ah,
                                    int M, int kshift) {
  int tid = blockIdx.x * 256 + threadIdx.x;          // over M*32
  if (tid >= M * 32) return;
  int m = tid >> 5, c = tid & 31;
  int b = m >> (12 + kshift);
  int idx = nidx[m];
  Ah[(size_t)m * 32 + c] = (_Float16)x[((size_t)b * DIN + c) * N0_PTS + idx];
}

__global__ void convert_w_kernel(const float* __restrict__ W,
                                 _Float16* __restrict__ Wh, int n) {
  int tid = blockIdx.x * 256 + threadIdx.x;
  if (tid < n) Wh[tid] = (_Float16)W[tid];
}

// ---------------------------------------------------------------------------
// 5) GEMM: D[16 samples x 16 couts] = X(16xK) * W^T(Kx16), f16 in / f32 acc
//    via v_wmma_f32_16x16x32_f16. CIN/COUT are template params so every loop
//    fully unrolls and the A fragments stay in statically-indexed VGPRs
//    (no v_movrels). Also emits deterministic per-block per-channel
//    sum/sumsq partials for the batchnorm.
// ---------------------------------------------------------------------------
template <int CIN, int COUT>
__global__ __launch_bounds__(256) void gemm_kernel(
    const _Float16* __restrict__ Xh, const _Float16* __restrict__ Wh,
    _Float16* __restrict__ Gh, float* __restrict__ pS, float* __restrict__ pQ) {
  constexpr int NKT = CIN >> 5;     // 1 or 2
  constexpr int NCT = COUT >> 4;    // 2,4,8
  __shared__ float lsum[8 * COUT], lsq[8 * COUT];
  const int lane  = threadIdx.x & 31;
  const int w     = threadIdx.x >> 5;
  const int mbase = (blockIdx.x * 8 + w) * 16;
  const int khalf = lane >> 4;

  // A matrix (16x32 MxK, f16): lane L holds row (L&15);
  // element e -> K = khalf*16 + e
  v16h a[NKT];
  const int mrowA = mbase + (lane & 15);
#pragma unroll
  for (int kt = 0; kt < NKT; ++kt) {
    const _Float16* pa = Xh + (size_t)mrowA * CIN + kt * 32 + khalf * 16;
    v8h lo = *reinterpret_cast<const v8h*>(pa);
    v8h hi = *reinterpret_cast<const v8h*>(pa + 8);
    a[kt] = __builtin_shufflevector(lo, hi, 0, 1, 2, 3, 4, 5, 6, 7,
                                    8, 9, 10, 11, 12, 13, 14, 15);
  }

#pragma unroll
  for (int ct = 0; ct < NCT; ++ct) {
    const int n = ct * 16 + (lane & 15);             // cout column
    v8f acc = {0.f, 0.f, 0.f, 0.f, 0.f, 0.f, 0.f, 0.f};
#pragma unroll
    for (int kt = 0; kt < NKT; ++kt) {
      // B matrix (32x16 KxN): lane L holds col (L&15); element e -> K = khalf*16+e
      v16h bm = *reinterpret_cast<const v16h*>(Wh + (size_t)n * CIN + kt * 32 + khalf * 16);
      acc = __builtin_amdgcn_wmma_f32_16x16x32_f16(
          false, a[kt], false, bm, (short)0, acc, false, false);
    }
    // D layout: VGPR r, lane L -> row = r + (L>>4)*8, col = L&15
    float s1 = 0.f, s2 = 0.f;
    const int mrowD = mbase + (khalf << 3);
#pragma unroll
    for (int r = 0; r < 8; ++r) {
      float v = acc[r];
      s1 += v; s2 += v * v;
      Gh[(size_t)(mrowD + r) * COUT + ct * 16 + (lane & 15)] = (_Float16)v;
    }
    s1 += __shfl_down(s1, 16, 32);
    s2 += __shfl_down(s2, 16, 32);
    if (lane < 16) {
      lsum[w * COUT + ct * 16 + lane] = s1;
      lsq [w * COUT + ct * 16 + lane] = s2;
    }
  }
  __syncthreads();
#pragma unroll 1
  for (int c = threadIdx.x; c < COUT; c += 256) {
    float s1 = 0.f, s2 = 0.f;
#pragma unroll
    for (int ww = 0; ww < 8; ++ww) { s1 += lsum[ww * COUT + c]; s2 += lsq[ww * COUT + c]; }
    pS[(size_t)blockIdx.x * COUT + c] = s1;
    pQ[(size_t)blockIdx.x * COUT + c] = s2;
  }
}

// ---------------------------------------------------------------------------
// 6) Finalize BN stats: per-channel deterministic tree reduce of partials,
//    emit scale = gamma*rsqrt(var+eps), shift = beta - mu*scale.
// ---------------------------------------------------------------------------
__global__ __launch_bounds__(256) void stats_kernel(
    const float* __restrict__ pS, const float* __restrict__ pQ,
    const float* __restrict__ gamma, const float* __restrict__ beta,
    float* __restrict__ scale, float* __restrict__ shift,
    int nblocks, int COUT, float invM) {
  __shared__ float a1[256], a2[256];
  const int c = blockIdx.x;
  float s1 = 0.f, s2 = 0.f;
  for (int i = threadIdx.x; i < nblocks; i += 256) {
    s1 += pS[(size_t)i * COUT + c];
    s2 += pQ[(size_t)i * COUT + c];
  }
  a1[threadIdx.x] = s1; a2[threadIdx.x] = s2;
  __syncthreads();
  for (int off = 128; off; off >>= 1) {
    if (threadIdx.x < off) {
      a1[threadIdx.x] += a1[threadIdx.x + off];
      a2[threadIdx.x] += a2[threadIdx.x + off];
    }
    __syncthreads();
  }
  if (threadIdx.x == 0) {
    float mu  = a1[0] * invM;
    float var = a2[0] * invM - mu * mu;
    float sc  = gamma[c] * rsqrtf(var + 1e-3f);
    scale[c] = sc;
    shift[c] = beta[c] - mu * sc;
  }
}

// ---------------------------------------------------------------------------
// 7) BN + leaky-relu, write next-layer f16 activations
// ---------------------------------------------------------------------------
__global__ void bn_apply_kernel(const _Float16* __restrict__ Gh,
                                const float* __restrict__ scale,
                                const float* __restrict__ shift,
                                _Float16* __restrict__ Oh, int total, int cmask) {
  int tid = blockIdx.x * 256 + threadIdx.x;
  if (tid >= total) return;
  int c = tid & cmask;
  float y = (float)Gh[tid] * scale[c] + shift[c];
  y = (y > 0.f) ? y : 0.01f * y;
  Oh[tid] = (_Float16)y;
}

// ---------------------------------------------------------------------------
// 8) Final layer: BN + leaky-relu + max over k, write d_out features
// ---------------------------------------------------------------------------
__global__ void bn_max_kernel(const _Float16* __restrict__ Gh,
                              const float* __restrict__ scale,
                              const float* __restrict__ shift,
                              float* __restrict__ out,
                              int k, int COUT, int lc, int coff) {
  int tid = blockIdx.x * 256 + threadIdx.x;          // over B*COUT*S
  int s = tid & (SPOINT - 1);
  int c = (tid >> 12) & (COUT - 1);
  int b = tid >> (12 + lc);
  float sc = scale[c], sh = shift[c];
  size_t m0 = ((size_t)((b << 12) + s)) * k;
  float mx = -3.4e38f;
  for (int j = 0; j < k; ++j) {
    float y = (float)Gh[(m0 + j) * COUT + c] * sc + sh;
    y = (y > 0.f) ? y : 0.01f * y;
    mx = fmaxf(mx, y);
  }
  out[(size_t)(BATCH * 3 * SPOINT) + ((size_t)(b * 192 + coff + c) << 12) + s] = mx;
}

// ---------------------------------------------------------------------------
extern "C" void kernel_launch(void* const* d_in, const int* in_sizes, int n_in,
                              void* d_out, int out_size, void* d_ws, size_t ws_size,
                              hipStream_t stream) {
  const float* qp = (const float*)d_in[0];
  const float* sp = (const float*)d_in[1];
  const float* x  = (const float*)d_in[2];
  // params0 leaves: d_in[3..14] (W,b,gamma,beta) x3 ; params1: d_in[15..26]
  float* out = (float*)d_out;

  // ---- workspace carve-up (256B aligned) ----
  char* base = (char*)d_ws; size_t off = 0;
  auto carve = [&](size_t bytes) -> void* {
    off = (off + 255) & ~(size_t)255;
    void* p = base + off; off += bytes; return p;
  };
  const size_t Mmax = (size_t)BATCH * SPOINT * 32;   // 524288
  int*       fpsIdx = (int*)carve((size_t)BATCH * SPOINT * 4);
  int*       nidx0  = (int*)carve((size_t)BATCH * SPOINT * 16 * 4);
  int*       nidx1  = (int*)carve((size_t)BATCH * SPOINT * 32 * 4);
  _Float16*  Wh     = (_Float16*)carve(128 * 64 * 2);
  float*     scale  = (float*)carve(128 * 4);
  float*     shift  = (float*)carve(128 * 4);
  float*     pS     = (float*)carve((Mmax / 128) * 128 * 4);
  float*     pQ     = (float*)carve((Mmax / 128) * 128 * 4);
  _Float16*  Ah     = (_Float16*)carve(Mmax * 64 * 2);
  _Float16*  Bh     = (_Float16*)carve(Mmax * 32 * 2);
  _Float16*  Gh     = (_Float16*)carve(Mmax * 128 * 2);

  // 1) FPS + xyz gather (xyz goes straight into d_out)
  fps_kernel<<<BATCH, 1024, 0, stream>>>(qp, fpsIdx);
  gather_xyz_kernel<<<(BATCH * SPOINT) / 256, 256, 0, stream>>>(qp, fpsIdx, out);

  const float radii[2]  = {0.2f, 0.4f};
  const int   ks[2]     = {16, 32};
  const int   kshift[2] = {4, 5};
  const int   cins[2][3]  = {{32, 32, 32}, {32, 32, 64}};
  const int   couts[2][3] = {{32, 32, 64}, {32, 64, 128}};
  const int   coffs[2]  = {0, 64};
  int*        nidxs[2]  = {nidx0, nidx1};

  for (int br = 0; br < 2; ++br) {
    const int k = ks[br];
    const int M = BATCH * SPOINT * k;
    int* nidx = nidxs[br];

    ball_query_kernel<<<(BATCH * SPOINT) / 8, 256, 0, stream>>>(
        sp, out, nidx, radii[br] * radii[br], k);
    gather_feats_kernel<<<(M * 32 + 255) / 256, 256, 0, stream>>>(
        x, nidx, Ah, M, kshift[br]);

    _Float16* ins[3] = {Ah, Bh, Ah};
    _Float16* bnout[3] = {Bh, Ah, nullptr};
    for (int l = 0; l < 3; ++l) {
      const int cin = cins[br][l], cout = couts[br][l];
      const int pbase = 3 + br * 12 + l * 4;
      const float* Wf    = (const float*)d_in[pbase + 0];
      const float* gamma = (const float*)d_in[pbase + 2];
      const float* beta  = (const float*)d_in[pbase + 3];

      convert_w_kernel<<<(cout * cin + 255) / 256, 256, 0, stream>>>(Wf, Wh, cout * cin);
      const int nblocks = M / 128;
      // template-dispatched GEMM (3 shapes exist)
      if (cin == 32 && cout == 32) {
        gemm_kernel<32, 32><<<nblocks, 256, 0, stream>>>(ins[l], Wh, Gh, pS, pQ);
      } else if (cin == 32 && cout == 64) {
        gemm_kernel<32, 64><<<nblocks, 256, 0, stream>>>(ins[l], Wh, Gh, pS, pQ);
      } else {
        gemm_kernel<64, 128><<<nblocks, 256, 0, stream>>>(ins[l], Wh, Gh, pS, pQ);
      }
      stats_kernel<<<cout, 256, 0, stream>>>(pS, pQ, gamma, beta, scale, shift,
                                             nblocks, cout, 1.0f / (float)M);
      if (l < 2) {
        bn_apply_kernel<<<(M * cout + 255) / 256, 256, 0, stream>>>(
            Gh, scale, shift, bnout[l], M * cout, cout - 1);
      } else {
        const int lc = (cout == 64) ? 6 : 7;
        bn_max_kernel<<<(BATCH * cout * SPOINT) / 256, 256, 0, stream>>>(
            Gh, scale, shift, out, k, cout, lc, coffs[br]);
      }
    }
  }
  (void)in_sizes; (void)n_in; (void)out_size; (void)ws_size;
}
